// HeteroGNN_75625784148346
// MI455X (gfx1250) — matrix operand order repaired and checked by reading, not compiled, not run on hardware
//
#include <hip/hip_runtime.h>

typedef __attribute__((ext_vector_type(16))) _Float16 v16h;
typedef __attribute__((ext_vector_type(8)))  float    v8f;

#define HDIM 64
#define ODIM 32
#define LNUM 2

// ---------------- conversion / fill ----------------

__global__ void f32_to_f16_kernel(const float* __restrict__ src, _Float16* __restrict__ dst, int n) {
    int i = blockIdx.x * blockDim.x + threadIdx.x;
    if (i < n) dst[i] = (_Float16)src[i];
}

__global__ void fill_kernel(float* __restrict__ p, float v, int n) {
    int i = blockIdx.x * blockDim.x + threadIdx.x;
    if (i < n) p[i] = v;
}

// ---------------- W pre-pack into WMMA B-fragment layout ----------------
// One block (256 threads) per 64x64 matrix. Output: 2048 u32 per matrix where
// u32 index = ((kc*4 + t)*32 + lane)*8 + i holds the f16 pair
// ( B[kc*32 + (lane>>4)*16 + 2i][t*16 + (lane&15)],  B[same k + 1][same n] ).
// This lets the GEMM read each B fragment as one aligned 32-byte LDS vector.

__global__ __launch_bounds__(256) void pack_w_kernel(const float* __restrict__ W,
                                                     unsigned int* __restrict__ out) {
    const int tid  = threadIdx.x;
    const int lane = tid & 31;
    const int cmb  = tid >> 5;           // (kc*4 + t), 0..7
    const int kc   = cmb >> 2;
    const int t    = cmb & 3;
    const size_t mb = (size_t)blockIdx.x * HDIM * HDIM;      // matrix base (f32)
    const size_t ob = (size_t)blockIdx.x * 2048;             // output base (u32)

    const int n = t * 16 + (lane & 15);
    #pragma unroll
    for (int i = 0; i < 8; ++i) {
        int k = kc * 32 + (lane >> 4) * 16 + 2 * i;
        union { _Float16 h[2]; unsigned int u; } p;
        p.h[0] = (_Float16)W[mb + (size_t)k * HDIM + n];
        p.h[1] = (_Float16)W[mb + (size_t)(k + 1) * HDIM + n];
        out[ob + (size_t)tid * 8 + i] = p.u;
    }
}

// ---------------- WMMA GEMM: Hout[Ns,64] = Xh[Ns,64] @ W[64,64] ----------------
// Block = 256 threads = 8 waves; each wave computes 16 rows x 64 cols via
// 8x v_wmma_f32_16x16x32_f16 (2 K-chunks x 4 N-tiles). Packed W + X tile in LDS.

__global__ __launch_bounds__(256) void gemm64_kernel(const _Float16* __restrict__ Xh,
                                                     const unsigned int* __restrict__ Wp,
                                                     float* __restrict__ Hout, int Ns) {
    __shared__ __align__(32) unsigned int sWp[2048];   // 8 KB, packed B fragments
    __shared__ __align__(16) _Float16     sX[128 * HDIM]; // 16 KB
    const int tid  = threadIdx.x;
    const int row0 = blockIdx.x * 128;

    // stage packed W
    #pragma unroll
    for (int it = 0; it < 8; ++it) sWp[tid + it * 256] = Wp[tid + it * 256];

    // stage X tile (128 rows x 64 f16), zero rows past Ns
    {
        const unsigned int* xg  = (const unsigned int*)Xh;
        unsigned int*       sXu = (unsigned int*)sX;
        #pragma unroll
        for (int it = 0; it < 16; ++it) {
            int idx = tid + it * 256;          // u32 index in tile, 32 per row
            int r   = row0 + (idx >> 5);
            sXu[idx] = (r < Ns) ? xg[(size_t)r * 32 + (idx & 31)] : 0u;
        }
    }
    __syncthreads();

    const int wave  = tid >> 5;
    const int lane  = tid & 31;
    const int m     = lane & 15;
    const int hsel  = lane >> 4;
    const int rbase = wave * 16;               // row-in-tile base for this wave

    v8f c[4] = {v8f{}, v8f{}, v8f{}, v8f{}};

    #pragma unroll
    for (int kc = 0; kc < 2; ++kc) {
        // A fragment (16x32 f16): lane m = row; half-wave selects K half (16 contiguous k).
        // Elements are LDS-contiguous in one 32B chunk.
        v16h a;
        #pragma unroll
        for (int i = 0; i < 8; ++i) {
            int k = kc * 32 + hsel * 16 + 2 * i;
            a[2 * i]     = sX[(rbase + m) * HDIM + k];
            a[2 * i + 1] = sX[(rbase + m) * HDIM + k + 1];
        }
        #pragma unroll
        for (int t = 0; t < 4; ++t) {
            const v16h b = *(const v16h*)&sWp[(kc * 4 + t) * 256 + lane * 8];
            c[t] = __builtin_amdgcn_wmma_f32_16x16x32_f16(false, a, false, b,
                                                          (short)0, c[t], false, false);
        }
    }

    // C layout: VGPR j -> row (j + hsel*8), N = t*16 + m
    if (row0 + 128 <= Ns) {                    // fast path: no per-element guards
        #pragma unroll
        for (int t = 0; t < 4; ++t) {
            #pragma unroll
            for (int j = 0; j < 8; ++j) {
                int r = row0 + rbase + j + hsel * 8;
                Hout[(size_t)r * HDIM + t * 16 + m] = c[t][j];
            }
        }
    } else {
        #pragma unroll
        for (int t = 0; t < 4; ++t) {
            #pragma unroll
            for (int j = 0; j < 8; ++j) {
                int r = row0 + rbase + j + hsel * 8;
                if (r < Ns) Hout[(size_t)r * HDIM + t * 16 + m] = c[t][j];
            }
        }
    }
}

// ---------------- per-row dot with attention vector ----------------

__global__ void rowdot_kernel(const float* __restrict__ h, const float* __restrict__ avec,
                              float* __restrict__ out, int N) {
    int row  = blockIdx.x * (blockDim.x >> 5) + (threadIdx.x >> 5);
    int lane = threadIdx.x & 31;
    if (row >= N) return;
    float s = h[(size_t)row * HDIM + lane] * avec[lane]
            + h[(size_t)row * HDIM + 32 + lane] * avec[32 + lane];
    #pragma unroll
    for (int off = 16; off > 0; off >>= 1) s += __shfl_xor(s, off, 32);
    if (lane == 0) out[row] = s;
}

// ---------------- edge kernels (segment softmax + scatter) ----------------

__device__ __forceinline__ void atomicMaxF(float* addr, float v) {
    if (v >= 0.f) atomicMax((int*)addr, __float_as_int(v));
    else          atomicMin((unsigned int*)addr, __float_as_uint(v));
}

__global__ void edge_score_kernel(const int* __restrict__ esrc, const int* __restrict__ edst,
                                  const float* __restrict__ ss, const float* __restrict__ sd,
                                  float* __restrict__ score, float* __restrict__ mmax, int E_) {
    int e = blockIdx.x * blockDim.x + threadIdx.x;
    if (e >= E_) return;
    int d   = edst[e];
    float v = ss[esrc[e]] + sd[d];
    v = (v > 0.f) ? v : 0.2f * v;            // leaky_relu(0.2)
    score[e] = v;
    atomicMaxF(&mmax[d], v);
}

__global__ void edge_exp_kernel(const int* __restrict__ edst, float* __restrict__ score,
                                const float* __restrict__ mmax, float* __restrict__ esum, int E_) {
    int e = blockIdx.x * blockDim.x + threadIdx.x;
    if (e >= E_) return;
    int d    = edst[e];
    float ex = __expf(score[e] - mmax[d]);
    score[e] = ex;
    atomicAdd(&esum[d], ex);
}

// 16 lanes per edge, each lane handles 4 contiguous floats (float4 gather, 4 atomics)
__global__ void scatter_kernel(const int* __restrict__ esrc, const int* __restrict__ edst,
                               const float* __restrict__ score, const float* __restrict__ esum,
                               const float* __restrict__ hs, float* __restrict__ acc, int E_) {
    int g   = blockIdx.x * blockDim.x + threadIdx.x;
    int e   = g >> 4;
    int l16 = g & 15;
    if (e >= E_) return;
    int s = esrc[e], d = edst[e];
    float alpha = score[e] / (esum[d] + 1e-16f);
    const float4 hv = *(const float4*)(hs + (size_t)s * HDIM + l16 * 4);
    float* ap = acc + (size_t)d * HDIM + l16 * 4;
    atomicAdd(ap + 0, hv.x * alpha);
    atomicAdd(ap + 1, hv.y * alpha);
    atomicAdd(ap + 2, hv.z * alpha);
    atomicAdd(ap + 3, hv.w * alpha);
}

// ---------------- bias + relu -> f16 state ----------------

__global__ void bias_relu_kernel(const float* __restrict__ acc, const float* __restrict__ bias,
                                 _Float16* __restrict__ xh, int n) {
    int i = blockIdx.x * blockDim.x + threadIdx.x;
    if (i >= n) return;
    float v = acc[i] + bias[i & (HDIM - 1)];
    xh[i] = (_Float16)(v > 0.f ? v : 0.f);
}

// ---------------- final MLP: relu(x@W1+b1)@W2+b2, one warp per row ----------------

__global__ void mlp_kernel(const _Float16* __restrict__ xh, const float* __restrict__ w1,
                           const float* __restrict__ b1, const float* __restrict__ w2,
                           const float* __restrict__ b2, float* __restrict__ out, int N) {
    int row  = blockIdx.x * (blockDim.x >> 5) + (threadIdx.x >> 5);
    int lane = threadIdx.x & 31;
    if (row >= N) return;
    float acc = 0.f;
    #pragma unroll 8
    for (int k = 0; k < HDIM; ++k) {
        float xv = (float)xh[(size_t)row * HDIM + k];
        acc += xv * w1[k * ODIM + lane];
    }
    float h = acc + b1[lane];
    h = h > 0.f ? h : 0.f;
    float p = h * w2[lane];
    #pragma unroll
    for (int off = 16; off > 0; off >>= 1) p += __shfl_xor(p, off, 32);
    if (lane == 0) out[row] = p + b2[0];
}

// ---------------- host orchestration ----------------

extern "C" void kernel_launch(void* const* d_in, const int* in_sizes, int n_in,
                              void* d_out, int out_size, void* d_ws, size_t ws_size,
                              hipStream_t stream) {
    (void)n_in; (void)out_size; (void)ws_size;

    const float* x_inst  = (const float*)d_in[0];
    const float* x_net   = (const float*)d_in[1];
    const int*   e_i2n   = (const int*)d_in[2];
    const int*   e_n2i   = (const int*)d_in[3];
    const float* W_src   = (const float*)d_in[4];
    const float* W_dst   = (const float*)d_in[5];
    const float* att_src = (const float*)d_in[6];
    const float* att_dst = (const float*)d_in[7];
    const float* bias_g  = (const float*)d_in[8];
    const float* lin1_w  = (const float*)d_in[9];
    const float* lin1_b  = (const float*)d_in[10];
    const float* lin2_w  = (const float*)d_in[11];
    const float* lin2_b  = (const float*)d_in[12];

    const int Ni   = in_sizes[0] / HDIM;
    const int Nn   = in_sizes[1] / HDIM;
    const int E_   = in_sizes[2] / 2;
    const int Nmax = (Ni > Nn) ? Ni : Nn;

    // workspace carve-out (256B-aligned)
    char*  ws  = (char*)d_ws;
    size_t off = 0;
    auto alloc = [&](size_t bytes) -> char* {
        char* p = ws + off;
        off = (off + bytes + 255) & ~(size_t)255;
        return p;
    };
    _Float16*     xh_inst = (_Float16*)alloc((size_t)Ni * HDIM * 2);
    _Float16*     xh_net  = (_Float16*)alloc((size_t)Nn * HDIM * 2);
    float*        hs      = (float*)alloc((size_t)Nmax * HDIM * 4);
    float*        hd      = (float*)alloc((size_t)Nmax * HDIM * 4);
    float*        acc_net = (float*)alloc((size_t)Nn * HDIM * 4);
    float*        acc_ins = (float*)alloc((size_t)Ni * HDIM * 4);
    float*        score   = (float*)alloc((size_t)E_ * 4);
    float*        ss      = (float*)alloc((size_t)Nmax * 4);
    float*        sd      = (float*)alloc((size_t)Nmax * 4);
    float*        mmax    = (float*)alloc((size_t)Nmax * 4);
    float*        esum    = (float*)alloc((size_t)Nmax * 4);
    unsigned int* wsp     = (unsigned int*)alloc((size_t)LNUM * 2 * 2048 * 4);
    unsigned int* wdp     = (unsigned int*)alloc((size_t)LNUM * 2 * 2048 * 4);

    // initial conversions / weight packing
    f32_to_f16_kernel<<<(Ni * HDIM + 255) / 256, 256, 0, stream>>>(x_inst, xh_inst, Ni * HDIM);
    f32_to_f16_kernel<<<(Nn * HDIM + 255) / 256, 256, 0, stream>>>(x_net,  xh_net,  Nn * HDIM);
    pack_w_kernel<<<LNUM * 2, 256, 0, stream>>>(W_src, wsp);
    pack_w_kernel<<<LNUM * 2, 256, 0, stream>>>(W_dst, wdp);

    auto run_conv = [&](const _Float16* srcH, int Ns, const _Float16* dstH, int Nd,
                        const int* esrc, const int* edst, int l, int r, float* accDst) {
        const unsigned int* WsP = wsp + (size_t)(l * 2 + r) * 2048;
        const unsigned int* WdP = wdp + (size_t)(l * 2 + r) * 2048;
        const float*        as_ = att_src + (size_t)(l * 2 + r) * HDIM;
        const float*        ad_ = att_dst + (size_t)(l * 2 + r) * HDIM;

        gemm64_kernel<<<(Ns + 127) / 128, 256, 0, stream>>>(srcH, WsP, hs, Ns);
        gemm64_kernel<<<(Nd + 127) / 128, 256, 0, stream>>>(dstH, WdP, hd, Nd);
        rowdot_kernel<<<(Ns + 7) / 8, 256, 0, stream>>>(hs, as_, ss, Ns);
        rowdot_kernel<<<(Nd + 7) / 8, 256, 0, stream>>>(hd, ad_, sd, Nd);
        fill_kernel<<<(Nd + 255) / 256, 256, 0, stream>>>(mmax, -1e30f, Nd);
        fill_kernel<<<(Nd + 255) / 256, 256, 0, stream>>>(esum, 0.f, Nd);
        fill_kernel<<<(Nd * HDIM + 255) / 256, 256, 0, stream>>>(accDst, 0.f, Nd * HDIM);
        edge_score_kernel<<<(E_ + 255) / 256, 256, 0, stream>>>(esrc, edst, ss, sd, score, mmax, E_);
        edge_exp_kernel<<<(E_ + 255) / 256, 256, 0, stream>>>(edst, score, mmax, esum, E_);
        scatter_kernel<<<((size_t)E_ * 16 + 255) / 256, 256, 0, stream>>>(esrc, edst, score, esum, hs, accDst, E_);
    };

    for (int l = 0; l < LNUM; ++l) {
        // relation 0: inst -> net (dst = net)
        run_conv(xh_inst, Ni, xh_net, Nn, e_i2n, e_i2n + E_, l, 0, acc_net);
        // relation 1: net -> inst (dst = inst)
        run_conv(xh_net, Nn, xh_inst, Ni, e_n2i, e_n2i + E_, l, 1, acc_ins);
        // update state (must happen after both convs read old state)
        bias_relu_kernel<<<(Nn * HDIM + 255) / 256, 256, 0, stream>>>(
            acc_net, bias_g + (size_t)(l * 2 + 0) * HDIM, xh_net, Nn * HDIM);
        bias_relu_kernel<<<(Ni * HDIM + 255) / 256, 256, 0, stream>>>(
            acc_ins, bias_g + (size_t)(l * 2 + 1) * HDIM, xh_inst, Ni * HDIM);
    }

    mlp_kernel<<<(Nn + 7) / 8, 256, 0, stream>>>(xh_net, lin1_w, lin1_b, lin2_w, lin2_b,
                                                 (float*)d_out, Nn);
}
